// AssociativeIncrementalAttention_51118700757276
// MI455X (gfx1250) — matrix operand order repaired and checked by reading, not compiled, not run on hardware
//
#include <hip/hip_runtime.h>
#include <hip/hip_bf16.h>

// ---- problem constants (match reference) ----
#define B_  2
#define S_  2048
#define D_  512
#define H_  8
#define HD_ 64
#define CONCEPT_BIAS 0.5f

typedef _Float16 v8h  __attribute__((ext_vector_type(8)));
typedef _Float16 v16h __attribute__((ext_vector_type(16)));
typedef float    v8f  __attribute__((ext_vector_type(8)));
typedef unsigned int v4u __attribute__((ext_vector_type(4)));
typedef int      v4i  __attribute__((ext_vector_type(4)));
typedef int      v8i  __attribute__((ext_vector_type(8)));

__device__ __forceinline__ v16h combine16(v8h lo, v8h hi) {
    v16h r;
#pragma unroll
    for (int j = 0; j < 8; ++j) { r[j] = lo[j]; r[j + 8] = hi[j]; }
    return r;
}

__device__ __forceinline__ v8f wmma_f16(v16h a, v16h b, v8f c) {
    // D = A(16x32 f16) x B(32x16 f16) + C(16x16 f32)
    return __builtin_amdgcn_wmma_f32_16x16x32_f16(
        /*neg_a=*/false, a, /*neg_b=*/false, b,
        /*c_mod=*/(short)0, c, /*reuse_a=*/false, /*reuse_b=*/false);
}

// -------------------------------------------------------------------------
// TDM: 2-D f16 tile load Global -> LDS via Tensor Data Mover.
// Descriptor packing per CDNA5 ISA (D# groups 0/1).
// -------------------------------------------------------------------------
__device__ __forceinline__ void tdm_load_2d_f16(
        unsigned lds_off, const void* gptr,
        unsigned tensor_d0, unsigned tensor_d1,
        unsigned tile_d0, unsigned tile_d1, unsigned stride0) {
    unsigned long long ga = (unsigned long long)gptr;
    v4u g0;
    g0[0] = 1u;                                             // count=1, user mode
    g0[1] = lds_off;                                        // lds_addr (bytes)
    g0[2] = (unsigned)ga;                                   // global_addr[31:0]
    g0[3] = (unsigned)((ga >> 32) & 0x01FFFFFFu) | (2u << 30);  // addr[56:32] | type=2
    v8i g1;
    g1[0] = (int)(1u << 16);                                // data_size=1 (2 bytes)
    g1[1] = (int)(tensor_d0 << 16);                         // tensor_dim0[15:0]
    g1[2] = (int)((tensor_d0 >> 16) | (tensor_d1 << 16));   // dim0[31:16] | dim1[15:0]
    g1[3] = (int)((tensor_d1 >> 16) | (tile_d0 << 16));     // dim1[31:16] | tile_dim0
    g1[4] = (int)(tile_d1 & 0xffffu);                       // tile_dim1 | tile_dim2=0
    g1[5] = (int)stride0;                                   // tensor_dim0_stride[31:0]
    g1[6] = 0;
    g1[7] = 0;
    v4i z4 = {0, 0, 0, 0};
#if defined(__clang_major__) && (__clang_major__ >= 23)
    v8i z8 = {0, 0, 0, 0, 0, 0, 0, 0};
    __builtin_amdgcn_tensor_load_to_lds(g0, g1, z4, z4, z8, 0);
#else
    __builtin_amdgcn_tensor_load_to_lds(g0, g1, z4, z4, 0);
#endif
}

// -------------------------------------------------------------------------
// Kernel 0: one-shot f32 -> f16 conversion of x and the four weights.
// dst layout: [xh (2M) | Wq (256K) | Wk | Wv | Wo] halves.
// grid = 3072 x 256, 4 elems/thread.
// -------------------------------------------------------------------------
__global__ void __launch_bounds__(256)
cvt_kernel(const float* __restrict__ x,
           const float* __restrict__ Wq, const float* __restrict__ Wk,
           const float* __restrict__ Wv, const float* __restrict__ Wo,
           _Float16* __restrict__ dst) {
    const size_t e = (size_t)blockIdx.x * 1024 + threadIdx.x * 4;
    const float* src;
    size_t off;
    if (e < (size_t)B_ * S_ * D_) {
        src = x; off = e;
    } else {
        const size_t we = e - (size_t)B_ * S_ * D_;
        const size_t w = we >> 18;                 // / (512*512)
        off = we & ((1u << 18) - 1);
        src = (w == 0) ? Wq : (w == 1) ? Wk : (w == 2) ? Wv : Wo;
    }
    const float4 v = *(const float4*)(src + off);
    _Float16* d = dst + e;
    d[0] = (_Float16)v.x; d[1] = (_Float16)v.y;
    d[2] = (_Float16)v.z; d[3] = (_Float16)v.w;
}

// -------------------------------------------------------------------------
// Kernel 1: QKV projection.  out = x @ W.T + b  (f16 WMMA, f32 accum)
// grid = (M/16 = 256, 4, 3{q,k,v}), block = 256 (8 waves; wave -> n-tile)
// A-tile (16x512 f16) staged to LDS by the Tensor Data Mover.
// Q,K stored [B,H,S,HD]; V stored transposed [B,H,HD,S].
// -------------------------------------------------------------------------
__global__ void __launch_bounds__(256)
qkv_proj_kernel(const _Float16* __restrict__ xh,
                const _Float16* __restrict__ wh,   // [3][512][512] f16 (q,k,v)
                const float* __restrict__ bq, const float* __restrict__ bk,
                const float* __restrict__ bv,
                _Float16* __restrict__ qw, _Float16* __restrict__ kw,
                _Float16* __restrict__ vtw) {
    __shared__ __align__(16) _Float16 xt[16 * D_];

    const int tid = threadIdx.x;
    const int m0  = blockIdx.x * 16;       // row tile over B*S
    const int z   = blockIdx.z;            // 0=q 1=k 2=v
    const _Float16* W    = wh + (size_t)z * D_ * D_;
    const float*    bias = (z == 0) ? bq : (z == 1) ? bk : bv;

    // TDM: stage x tile (16 rows x 512 cols, f16) into LDS
    if (tid < 32) {
        tdm_load_2d_f16((unsigned)(unsigned long long)(const void*)xt,
                        xh + (size_t)m0 * D_,
                        /*tensor_d0=*/D_, /*tensor_d1=*/B_ * S_,
                        /*tile_d0=*/D_, /*tile_d1=*/16, /*stride0=*/D_);
        __builtin_amdgcn_s_wait_tensorcnt(0);
    }
    __syncthreads();

    const int lane = tid & 31, wave = tid >> 5;
    const int half = lane >> 4, l16 = lane & 15;
    const int aoff = half ? 8 : 0;
    const int n0   = (blockIdx.y * 8 + wave) * 16;

    v8f acc = {};
#pragma unroll 4
    for (int kk = 0; kk < D_; kk += 32) {
        const _Float16* ap = xt + l16 * D_ + kk + aoff;
        v16h A = combine16(*(const v8h*)ap, *(const v8h*)(ap + 16));
        const _Float16* wp = W + (size_t)(n0 + l16) * D_ + kk + (half ? 8 : 0);
        v16h Bf = combine16(*(const v8h*)wp, *(const v8h*)(wp + 16));
        acc = wmma_f16(A, Bf, acc);
    }

#pragma unroll
    for (int r = 0; r < 8; ++r) {
        const int m = m0 + r + half * 8;       // global row in B*S
        const int n = n0 + l16;                // global col in D
        const int b = m >> 11, s = m & (S_ - 1);
        const int h = n >> 6,  hd = n & (HD_ - 1);
        const _Float16 hv = (_Float16)(acc[r] + bias[n]);
        if (z == 0)
            qw[(((size_t)(b * H_ + h) * S_) + s) * HD_ + hd] = hv;
        else if (z == 1)
            kw[(((size_t)(b * H_ + h) * S_) + s) * HD_ + hd] = hv;
        else
            vtw[(((size_t)(b * H_ + h) * HD_) + hd) * S_ + s] = hv;
    }
}

// -------------------------------------------------------------------------
// Kernel 2: flash attention. grid = B*S/16 = 256 blocks, 256 thr (8 waves).
// block owns (b, 16 q rows); wave w = head w. Bias tile built once per
// k-step in LDS and reused by all 8 heads.
// -------------------------------------------------------------------------
__global__ void __launch_bounds__(256)
attn_kernel(const float* __restrict__ amask,
            const float* __restrict__ ipa,
            const float* __restrict__ assoc,
            const int* __restrict__ cid,
            const unsigned char* __restrict__ kpm,
            const _Float16* __restrict__ qw,
            const _Float16* __restrict__ kw,
            const _Float16* __restrict__ vtw,
            _Float16* __restrict__ ctx) {
    __shared__ __align__(16) float    btile[16 * 32];
    __shared__ __align__(16) _Float16 pst[8][16 * 32];
    __shared__ int cidq[16];

    const int tid  = threadIdx.x;
    const int lane = tid & 31;
    const int wave = tid >> 5;              // head index
    const int b    = blockIdx.x >> 7;
    const int q0   = (blockIdx.x & 127) << 4;

    const int half = lane >> 4, l16 = lane & 15;
    const int aoff = half ? 8 : 0;

    if (tid < 16) cidq[tid] = cid[b * S_ + q0 + tid];

    const size_t hb = (size_t)(b * H_ + wave) * S_ * HD_;
    const _Float16* Q  = qw  + hb;          // [S][HD]
    const _Float16* K  = kw  + hb;          // [S][HD]
    const _Float16* Vt = vtw + hb;          // [HD][S]

    // load Q fragments (16 rows x 64, two k-steps of 32)
    v16h Qa[2];
    {
        const int row = q0 + l16;
#pragma unroll
        for (int t = 0; t < 2; ++t) {
            const _Float16* p = Q + (size_t)row * HD_ + t * 32 + aoff;
            Qa[t] = combine16(*(const v8h*)p, *(const v8h*)(p + 16));
        }
    }

    float mst[8], lst[8];
    v8f O[4];
#pragma unroll
    for (int r = 0; r < 8; ++r) { mst[r] = -__builtin_inff(); lst[r] = 0.f; }
#pragma unroll
    for (int t = 0; t < 4; ++t) { v8f zz = {}; O[t] = zz; }

    const float scale = 0.125f;             // 1/sqrt(HD)

    // bias-fill coordinates for this thread: 2 contiguous elements
    const int e0  = tid * 2;
    const int fql = e0 >> 5;                // 0..15
    const int fkl = e0 & 31;                // even, 0..30
    const int fq  = q0 + fql;
    const int cq  = cid[b * S_ + fq];       // fine: L0 hit, uniform-ish

    for (int k0 = 0; k0 < S_; k0 += 32) {
        __syncthreads();
        // cooperative bias tile: [16 q][32 k], 2 contiguous elems/thread
        {
            const int k = k0 + fkl;
            const size_t rb = ((size_t)b * S_ + fq) * S_ + k;
            const float2 m2 = *(const float2*)&amask[(size_t)fq * S_ + k];
            const float2 i2 = *(const float2*)&ipa[rb];
            const float2 a2 = *(const float2*)&assoc[rb];
            // prefetch next iteration's bias lines (speculative, dropped if OOB)
            __builtin_prefetch(&ipa[rb + 32], 0, 0);
            __builtin_prefetch(&assoc[rb + 32], 0, 0);
            float b0 = m2.x + i2.x + a2.x;
            float b1 = m2.y + i2.y + a2.y;
            const int ck0 = cid[b * S_ + k];
            const int ck1 = cid[b * S_ + k + 1];
            if (cq == ck0 && cq >= 0 && fq != k)     b0 += CONCEPT_BIAS;
            if (cq == ck1 && cq >= 0 && fq != k + 1) b1 += CONCEPT_BIAS;
            if (kpm[b * S_ + k])     b0 = -__builtin_inff();
            if (kpm[b * S_ + k + 1]) b1 = -__builtin_inff();
            btile[e0]     = b0;
            btile[e0 + 1] = b1;
        }
        __syncthreads();

        // scores: S0 = keys k0..k0+15, S1 = keys k0+16..k0+31
        v8f S0 = {}, S1 = {};
#pragma unroll
        for (int hh = 0; hh < 2; ++hh) {
            {
                const _Float16* p =
                    K + (size_t)(k0 + l16) * HD_ + hh * 32 + (half ? 8 : 0);
                v16h Bf = combine16(*(const v8h*)p, *(const v8h*)(p + 16));
                S0 = wmma_f16(Qa[hh], Bf, S0);
            }
            {
                const _Float16* p =
                    K + (size_t)(k0 + 16 + l16) * HD_ + hh * 32 + (half ? 8 : 0);
                v16h Bf = combine16(*(const v8h*)p, *(const v8h*)(p + 16));
                S1 = wmma_f16(Qa[hh], Bf, S1);
            }
        }

        // online softmax per row
#pragma unroll
        for (int r = 0; r < 8; ++r) {
            const int m_l = r + half * 8;
            float s0 = S0[r] * scale + btile[m_l * 32 + l16];
            float s1 = S1[r] * scale + btile[m_l * 32 + 16 + l16];
            float rowm = fmaxf(s0, s1);
#pragma unroll
            for (int msk = 1; msk < 16; msk <<= 1)
                rowm = fmaxf(rowm, __shfl_xor(rowm, msk, 32));
            const float mnew = fmaxf(mst[r], rowm);
            const float p0 = __expf(s0 - mnew);
            const float p1 = __expf(s1 - mnew);
            float rs = p0 + p1;
#pragma unroll
            for (int msk = 1; msk < 16; msk <<= 1)
                rs += __shfl_xor(rs, msk, 32);
            const float corr = __expf(mst[r] - mnew);
            lst[r] = lst[r] * corr + rs;
            mst[r] = mnew;
#pragma unroll
            for (int t = 0; t < 4; ++t) O[t][r] *= corr;
            pst[wave][m_l * 32 + l16]      = (_Float16)p0;
            pst[wave][m_l * 32 + 16 + l16] = (_Float16)p1;
        }

        // reload P in WMMA A layout (per-wave stage; same-wave DS ordering)
        const _Float16* pp = &pst[wave][l16 * 32 + aoff];
        v16h Pa = combine16(*(const v8h*)pp, *(const v8h*)(pp + 16));

        // O += P(16x32) x V(32x64): 4 hd-tiles, Vt rows contiguous in s
#pragma unroll
        for (int t = 0; t < 4; ++t) {
            const _Float16* vp =
                Vt + (size_t)(t * 16 + l16) * S_ + k0 + (half ? 8 : 0);
            v16h Bf = combine16(*(const v8h*)vp, *(const v8h*)(vp + 16));
            O[t] = wmma_f16(Pa, Bf, O[t]);
        }
    }

    // epilogue: normalize, write ctx [B,S,D] f16
#pragma unroll
    for (int r = 0; r < 8; ++r) {
        const float inv = 1.0f / lst[r];
        const int srow = q0 + r + half * 8;
        const size_t base = ((size_t)b * S_ + srow) * D_ + wave * HD_;
#pragma unroll
        for (int t = 0; t < 4; ++t)
            ctx[base + t * 16 + l16] = (_Float16)(O[t][r] * inv);
    }
}

// -------------------------------------------------------------------------
// Kernel 3: output projection. out = ctx @ Wo.T + bo (f16 WMMA, f32 out)
// grid = (256, 4), block = 256. ctx tile staged via TDM.
// -------------------------------------------------------------------------
__global__ void __launch_bounds__(256)
oproj_kernel(const _Float16* __restrict__ ctx,
             const _Float16* __restrict__ woh,   // [512][512] f16
             const float* __restrict__ bo,
             float* __restrict__ out) {
    __shared__ __align__(16) _Float16 ct[16 * D_];

    const int tid = threadIdx.x;
    const int m0  = blockIdx.x * 16;

    if (tid < 32) {
        tdm_load_2d_f16((unsigned)(unsigned long long)(const void*)ct,
                        ctx + (size_t)m0 * D_,
                        /*tensor_d0=*/D_, /*tensor_d1=*/B_ * S_,
                        /*tile_d0=*/D_, /*tile_d1=*/16, /*stride0=*/D_);
        __builtin_amdgcn_s_wait_tensorcnt(0);
    }
    __syncthreads();

    const int lane = tid & 31, wave = tid >> 5;
    const int half = lane >> 4, l16 = lane & 15;
    const int aoff = half ? 8 : 0;
    const int n0   = (blockIdx.y * 8 + wave) * 16;

    v8f acc = {};
#pragma unroll 4
    for (int kk = 0; kk < D_; kk += 32) {
        const _Float16* ap = ct + l16 * D_ + kk + aoff;
        v16h A = combine16(*(const v8h*)ap, *(const v8h*)(ap + 16));
        const _Float16* wp = woh + (size_t)(n0 + l16) * D_ + kk + (half ? 8 : 0);
        v16h Bf = combine16(*(const v8h*)wp, *(const v8h*)(wp + 16));
        acc = wmma_f16(A, Bf, acc);
    }

#pragma unroll
    for (int r = 0; r < 8; ++r) {
        const int m = m0 + r + half * 8;
        const int n = n0 + l16;
        out[(size_t)m * D_ + n] = acc[r] + bo[n];
    }
}

// -------------------------------------------------------------------------
extern "C" void kernel_launch(void* const* d_in, const int* in_sizes, int n_in,
                              void* d_out, int out_size, void* d_ws, size_t ws_size,
                              hipStream_t stream) {
    const float* x     = (const float*)d_in[0];
    const float* Wq    = (const float*)d_in[1];
    const float* bq    = (const float*)d_in[2];
    const float* Wk    = (const float*)d_in[3];
    const float* bk    = (const float*)d_in[4];
    const float* Wv    = (const float*)d_in[5];
    const float* bv    = (const float*)d_in[6];
    const float* Wo    = (const float*)d_in[7];
    const float* bo    = (const float*)d_in[8];
    const float* amask = (const float*)d_in[9];
    const float* ipa   = (const float*)d_in[10];
    const float* assoc = (const float*)d_in[11];
    const int*   cid   = (const int*)d_in[12];
    const unsigned char* kpm = (const unsigned char*)d_in[13];
    float* out = (float*)d_out;

    const size_t NQ = (size_t)B_ * H_ * S_ * HD_;   // 2,097,152
    const size_t NW = (size_t)D_ * D_;              //   262,144
    _Float16* qw  = (_Float16*)d_ws;
    _Float16* kw  = qw + NQ;
    _Float16* vtw = kw + NQ;
    _Float16* ctx = vtw + NQ;
    _Float16* xh  = ctx + NQ;       // also base of cvt dst region
    _Float16* wh  = xh + NQ;        // [Wq|Wk|Wv|Wo] f16
    _Float16* woh = wh + 3 * NW;

    // 0) f32 -> f16 pre-pass (x + 4 weights)
    cvt_kernel<<<dim3((unsigned)((NQ + 4 * NW) / 1024)), 256, 0, stream>>>(
        x, Wq, Wk, Wv, Wo, xh);

    // 1) QKV projections (TDM-staged A tiles)
    dim3 g1((B_ * S_) / 16, 4, 3);
    qkv_proj_kernel<<<g1, 256, 0, stream>>>(xh, wh, bq, bk, bv, qw, kw, vtw);

    // 2) flash attention with fused biases
    dim3 g2((B_ * S_) / 16);
    attn_kernel<<<g2, 256, 0, stream>>>(amask, ipa, assoc, cid, kpm,
                                        qw, kw, vtw, ctx);

    // 3) output projection
    dim3 g3((B_ * S_) / 16, 4);
    oproj_kernel<<<g3, 256, 0, stream>>>(ctx, woh, bo, out);
}
